// SelfAttn_39135742001892
// MI455X (gfx1250) — matrix operand order repaired and checked by reading, not compiled, not run on hardware
//
#include <hip/hip_runtime.h>
#include <hip/hip_bf16.h>

// ---------------------------------------------------------------------------
// Self-attention (non-local block) B=4, C=256, H=W=64 (N=4096, DQK=32).
//   proj_wmma_kernel<MTILES> : q/k/v = oscale*(W @ x + b) (WMMA f16).
//        One workgroup computes ALL M rows for a 64-pixel slice; x chunk
//        staged once per k0 and reused by MTILES WMMAs. q,k stored
//        TRANSPOSED [N][32]; q pre-scaled by log2e/sqrt(C) so the attention
//        kernel's logits arrive already in the exp2 domain.
//   attn_flash_kernel : flash attention, 64-key tiles, V tiles staged with
//        CDNA5 async DMA (global_load_async_to_lds_b128, ASYNCcnt), raw
//        v_exp_f32 softmax, vectorized epilogue out = gamma*O + x.
// Workspace: qT (1MB) + kT (1MB) + v (8MB) f16 = 10 MB.
// ---------------------------------------------------------------------------

typedef __attribute__((ext_vector_type(16))) _Float16 v16h;
typedef __attribute__((ext_vector_type(8)))  _Float16 v8h;
typedef __attribute__((ext_vector_type(4)))  _Float16 v4h;
typedef __attribute__((ext_vector_type(8)))  float    v8f;
typedef __attribute__((ext_vector_type(4)))  float    v4f;

static constexpr int BATCH  = 4;
static constexpr int CDIM   = 256;
static constexpr int NSEQ   = 4096;  // 64*64
static constexpr int DQKD   = 32;    // C/8
static constexpr int PITCH  = 40;    // proj LDS pitch (halfs), 16B aligned
static constexpr int PITCHV = 72;    // attn LDS pitch for 64-key tiles

// ---- fast math (guarded raw-instruction builtins) -------------------------

__device__ __forceinline__ float fast_exp2(float x) {
#if __has_builtin(__builtin_amdgcn_exp2f)
  return __builtin_amdgcn_exp2f(x);   // raw v_exp_f32 (args <= 0 here)
#else
  return exp2f(x);
#endif
}

__device__ __forceinline__ float fast_rcp(float x) {
#if __has_builtin(__builtin_amdgcn_rcpf)
  return __builtin_amdgcn_rcpf(x);    // raw v_rcp_f32
#else
  return 1.0f / x;
#endif
}

// ---- CDNA5 async-copy helpers ---------------------------------------------

typedef __attribute__((address_space(3))) _Float16 lds_f16;

__device__ __forceinline__ unsigned lds_off_of(const _Float16* p) {
  // generic -> LDS addrspace cast, then ptrtoint = 32-bit LDS offset
  return (unsigned)(unsigned long long)(const lds_f16*)p;
}

__device__ __forceinline__ void async_copy_b128(const _Float16* gsrc, _Float16* ldst) {
  unsigned loff = lds_off_of(ldst);
  unsigned long long ga = (unsigned long long)(uintptr_t)gsrc;
  asm volatile("global_load_async_to_lds_b128 %0, %1, off"
               :: "v"(loff), "v"(ga) : "memory");
}

__device__ __forceinline__ void wait_async0() {
#if __has_builtin(__builtin_amdgcn_s_wait_asynccnt)
  __builtin_amdgcn_s_wait_asynccnt(0);
#else
  asm volatile("s_wait_asynccnt 0x0" ::: "memory");
#endif
}

// ---- WMMA fragment loaders (ISA 7.12.2 layouts, wave32) -------------------
// Work for both LDS and global pointers; ldh (halfs) must keep 16B alignment.

// A fragment: 16x32 f16 (M x K), row-major.
__device__ __forceinline__ v16h load_a_frag(const _Float16* a, int ldh) {
  const int lane = threadIdx.x & 31;
  const int m    = lane & 15;
  const int hi   = lane >> 4;
  const _Float16* row = a + m * ldh;
  v8h lo = *(const v8h*)(row + hi * 8);
  v8h hs = *(const v8h*)(row + 16 + hi * 8);
  v16h r;
#pragma unroll
  for (int i = 0; i < 8; ++i) { r[i] = lo[i]; r[8 + i] = hs[i]; }
  return r;
}

// B fragment: 32x16 f16 (K x N), stored K-transposed: bt[n*ldh + k].
__device__ __forceinline__ v16h load_b_frag(const _Float16* bt, int ldh) {
  const int lane = threadIdx.x & 31;
  const int n    = lane & 15;
  const int koff = (lane >> 4) * 8;
  const _Float16* p = bt + (size_t)n * ldh + koff;
  v8h x0 = *(const v8h*)(p);
  v8h x1 = *(const v8h*)(p + 16);
  v16h r;
#pragma unroll
  for (int i = 0; i < 8; ++i) { r[i] = x0[i]; r[8 + i] = x1[i]; }
  return r;
}

__device__ __forceinline__ v8f wmma_f16(v16h a, v16h b, v8f c) {
  return __builtin_amdgcn_wmma_f32_16x16x32_f16(
      false, a, false, b, (short)0, c, false, false);
}

// ---------------------------------------------------------------------------
// Projection: out[m,n] = oscale*(sum_c W[m,c] x[b,c,n] + bias[m]), M=MTILES*16.
// grid (NSEQ/64, 1, B), block 128 (4 waves); wave w owns n-offset 16w.
// transpose_out=0 -> out[b][m][N] (v); transpose_out=1 -> out[b][n][M] (q,k).
// ---------------------------------------------------------------------------
template <int MTILES>
__global__ __launch_bounds__(128)
void proj_wmma_kernel(const float* __restrict__ W, const float* __restrict__ bias,
                      const float* __restrict__ x, _Float16* __restrict__ out,
                      int transpose_out, float oscale) {
  constexpr int M = MTILES * 16;
  const int n0   = blockIdx.x * 64;
  const int b    = blockIdx.z;
  const int t    = threadIdx.x;
  const int w    = t >> 5;
  const int lane = t & 31;

  __shared__ __align__(16) _Float16 Wt[M * PITCH];
  __shared__ __align__(16) _Float16 Xt[64 * PITCH];

  const float* xb = x + (size_t)b * CDIM * NSEQ;
  v8f acc[MTILES];
  v8f zero = {};
#pragma unroll
  for (int mt = 0; mt < MTILES; ++mt) acc[mt] = zero;

  for (int k0 = 0; k0 < CDIM; k0 += 32) {
    // stage W chunk: M rows x 32 k, one float4 per task (M*8 tasks)
#pragma unroll
    for (int e = 0; e < MTILES; ++e) {
      const int idx = t + 128 * e;
      const int m = idx >> 3, g = idx & 7;
      v4f wv = *(const v4f*)&W[(size_t)m * CDIM + k0 + g * 4];
      v4h hv;
#pragma unroll
      for (int i = 0; i < 4; ++i) hv[i] = (_Float16)wv[i];
      *(v4h*)(Wt + m * PITCH + g * 4) = hv;
    }
    // stage x chunk transposed: Xt[n][k] = x[k0+k][n0+n]; float4 along n
#pragma unroll
    for (int e = 0; e < 4; ++e) {
      const int idx = t + 128 * e;      // 512 tasks: k in [0,32), g in [0,16)
      const int k = idx & 31, g = idx >> 5;
      v4f xv = *(const v4f*)&xb[(size_t)(k0 + k) * NSEQ + n0 + g * 4];
#pragma unroll
      for (int i = 0; i < 4; ++i)
        Xt[(g * 4 + i) * PITCH + k] = (_Float16)xv[i];
    }
    __syncthreads();
    const v16h bf = load_b_frag(Xt + (w * 16) * PITCH, PITCH);  // invariant over mt
#pragma unroll
    for (int mt = 0; mt < MTILES; ++mt) {
      v16h a = load_a_frag(Wt + mt * 16 * PITCH, PITCH);
      acc[mt] = wmma_f16(a, bf, acc[mt]);
    }
    __syncthreads();
  }

  // epilogue; C/D layout: VGPR v -> row v / v+8, col = lane%16
  const int rbase = (lane >> 4) * 8;
  const int col   = n0 + w * 16 + (lane & 15);
  _Float16* ob = out + (size_t)b * M * NSEQ;
#pragma unroll
  for (int mt = 0; mt < MTILES; ++mt) {
    const int m0 = mt * 16;
    if (transpose_out) {
      v8h hv;
#pragma unroll
      for (int v = 0; v < 8; ++v)
        hv[v] = (_Float16)((acc[mt][v] + bias[m0 + rbase + v]) * oscale);
      *(v8h*)(ob + (size_t)col * M + m0 + rbase) = hv;
    } else {
#pragma unroll
      for (int v = 0; v < 8; ++v) {
        int row = m0 + rbase + v;
        ob[(size_t)row * NSEQ + col] = (_Float16)((acc[mt][v] + bias[row]) * oscale);
      }
    }
  }
}

// ---------------------------------------------------------------------------
// Flash attention, 64-key tiles. grid (NSEQ/64, B), block 256 (8 waves).
// Wave w: query-row tile r = w/2 (16 queries), channel half h = w%2 (128 ch).
// V tile DMA'd to LDS asynchronously while S-WMMAs + softmax run.
// q is pre-scaled by log2e/sqrt(C) -> scores are exp2-domain logits already.
// ---------------------------------------------------------------------------
__global__ __launch_bounds__(256)
void attn_flash_kernel(const _Float16* __restrict__ qT,  // [B][N][32] (scaled)
                       const _Float16* __restrict__ kT,  // [B][N][32]
                       const _Float16* __restrict__ vg,  // [B][256][N]
                       const float* __restrict__ x,      // [B][256][N]
                       const float* __restrict__ gamma_p,
                       float* __restrict__ out) {
  const int q0   = blockIdx.x * 64;
  const int b    = blockIdx.y;
  const int t    = threadIdx.x;
  const int w    = t >> 5;
  const int lane = t & 31;
  const int r    = w >> 1;
  const int h    = w & 1;

  __shared__ __align__(16) _Float16 Vt[256 * PITCHV];     // Vt[c][j], 64 keys
  __shared__ __align__(16) _Float16 Pb[4][16 * PITCHV];   // P per row-tile

  const _Float16* qb = qT + (size_t)b * NSEQ * DQKD;
  const _Float16* kb = kT + (size_t)b * NSEQ * DQKD;
  const _Float16* vb = vg + (size_t)b * CDIM * NSEQ;

  // Loop-invariant Q fragment, direct from global (row stride 32 halfs).
  const v16h aQ = load_a_frag(qb + (size_t)(q0 + r * 16) * DQKD, DQKD);

  float mrow[8], lrow[8];
  v8f O[8];
  v8f zero = {};
#pragma unroll
  for (int v = 0; v < 8; ++v) { mrow[v] = -1e30f; lrow[v] = 0.f; O[v] = zero; }

  const int vc8 = t >> 3;   // V staging: channel row base (0..31)
  const int ch8 = t & 7;    // 8-half chunk within 64-key row

  for (int j0 = 0; j0 < NSEQ; j0 += 64) {
    // kick off async DMA of the V tile into LDS (overlaps S + softmax)
#pragma unroll
    for (int e = 0; e < 8; ++e) {
      const int c = 32 * e + vc8;
      async_copy_b128(vb + (size_t)c * NSEQ + j0 + ch8 * 8,
                      Vt + c * PITCHV + ch8 * 8);
    }
    // prefetch next K tile rows (global_prefetch_b8), speculative
    if (t < 64 && j0 + 64 < NSEQ)
      __builtin_prefetch(kb + (size_t)(j0 + 64 + t) * DQKD, 0, 0);

    // S = Q^T K : 16 queries x 64 keys, K fragments direct from global
    v8f s[4];
#pragma unroll
    for (int i = 0; i < 4; ++i)
      s[i] = wmma_f16(aQ, load_b_frag(kb + (size_t)(j0 + 16 * i) * DQKD, DQKD), zero);

    // online softmax (exp2 domain); width-16 shuffle reductions over columns
    float p[4][8], alpha[8];
#pragma unroll
    for (int v = 0; v < 8; ++v) {
      float a0 = s[0][v], a1 = s[1][v], a2 = s[2][v], a3 = s[3][v];
      float mx = fmaxf(fmaxf(a0, a1), fmaxf(a2, a3));
      mx = fmaxf(mx, __shfl_xor(mx, 1, 16));
      mx = fmaxf(mx, __shfl_xor(mx, 2, 16));
      mx = fmaxf(mx, __shfl_xor(mx, 4, 16));
      mx = fmaxf(mx, __shfl_xor(mx, 8, 16));
      float mn = fmaxf(mrow[v], mx);
      float al = fast_exp2(mrow[v] - mn);
      float e0 = fast_exp2(a0 - mn), e1 = fast_exp2(a1 - mn);
      float e2 = fast_exp2(a2 - mn), e3 = fast_exp2(a3 - mn);
      float rs = (e0 + e1) + (e2 + e3);
      rs += __shfl_xor(rs, 1, 16);
      rs += __shfl_xor(rs, 2, 16);
      rs += __shfl_xor(rs, 4, 16);
      rs += __shfl_xor(rs, 8, 16);
      lrow[v] = lrow[v] * al + rs;
      mrow[v] = mn;
      alpha[v] = al;
      p[0][v] = e0; p[1][v] = e1; p[2][v] = e2; p[3][v] = e3;
    }
#pragma unroll
    for (int tt = 0; tt < 8; ++tt)
#pragma unroll
      for (int v = 0; v < 8; ++v) O[tt][v] *= alpha[v];

    // stage P (f16) into A-matrix layout; one writer wave per row-tile
    if (h == 0) {
      const int row = (lane >> 4) * 8;
      const int cl  = lane & 15;
#pragma unroll
      for (int v = 0; v < 8; ++v) {
#pragma unroll
        for (int i = 0; i < 4; ++i)
          Pb[r][(row + v) * PITCHV + 16 * i + cl] = (_Float16)p[i][v];
      }
    }
    wait_async0();     // this wave's V DMA landed in LDS
    __syncthreads();   // all waves' V tile + P tiles visible

    // O += P @ V^T over this wave's 128 channels (two 32-key halves)
    v16h aP0 = load_a_frag(Pb[r], PITCHV);
    v16h aP1 = load_a_frag(Pb[r] + 32, PITCHV);
#pragma unroll
    for (int tt = 0; tt < 8; ++tt) {
      const _Float16* vrow = Vt + (size_t)(h * 128 + tt * 16) * PITCHV;
      O[tt] = wmma_f16(aP0, load_b_frag(vrow, PITCHV), O[tt]);
      O[tt] = wmma_f16(aP1, load_b_frag(vrow + 32, PITCHV), O[tt]);
    }
    __syncthreads();   // protect Vt/Pb before next iteration's stores
  }

  // epilogue: out = gamma * (O/l) + x.
  // Per lane, the 8 accumulator elements are 8 CONSECUTIVE queries at one
  // channel, and out[c][i] is i-contiguous -> vector loads/stores.
  const float gamma = gamma_p[0];
  const int rbase = (lane >> 4) * 8;
  const int cl    = lane & 15;
  const float* xb = x + (size_t)b * CDIM * NSEQ;
  float* ob       = out + (size_t)b * CDIM * NSEQ;
  float gi[8];
#pragma unroll
  for (int v = 0; v < 8; ++v) gi[v] = gamma * fast_rcp(lrow[v]);
#pragma unroll
  for (int tt = 0; tt < 8; ++tt) {
    const int c = h * 128 + tt * 16 + cl;
    const size_t base = (size_t)c * NSEQ + q0 + rbase;
    v4f x0 = *(const v4f*)(xb + base);
    v4f x1 = *(const v4f*)(xb + base + 4);
    v4f o0, o1;
#pragma unroll
    for (int v = 0; v < 4; ++v) {
      o0[v] = O[tt][v] * gi[v] + x0[v];
      o1[v] = O[tt][4 + v] * gi[4 + v] + x1[v];
    }
    *(v4f*)(ob + base)     = o0;
    *(v4f*)(ob + base + 4) = o1;
  }
}

// ---------------------------------------------------------------------------

extern "C" void kernel_launch(void* const* d_in, const int* in_sizes, int n_in,
                              void* d_out, int out_size, void* d_ws, size_t ws_size,
                              hipStream_t stream) {
  const float* x     = (const float*)d_in[0];
  const float* Wq    = (const float*)d_in[1];
  const float* bq    = (const float*)d_in[2];
  const float* Wk    = (const float*)d_in[3];
  const float* bk    = (const float*)d_in[4];
  const float* Wv    = (const float*)d_in[5];
  const float* bv    = (const float*)d_in[6];
  const float* gamma = (const float*)d_in[7];

  // workspace (f16): qT[B][N][32], kT[B][N][32], v[B][256][N] = 10 MB
  _Float16* qws = (_Float16*)d_ws;
  _Float16* kws = qws + (size_t)BATCH * NSEQ * DQKD;
  _Float16* vws = kws + (size_t)BATCH * NSEQ * DQKD;

  const float SC2 = 0.0625f * 1.4426950408889634f;  // log2e / sqrt(C)

  dim3 pblk(128);
  dim3 pgrid(NSEQ / 64, 1, BATCH);
  proj_wmma_kernel<2><<<pgrid, pblk, 0, stream>>>(Wq, bq, x, qws, 1, SC2);
  proj_wmma_kernel<2><<<pgrid, pblk, 0, stream>>>(Wk, bk, x, kws, 1, 1.0f);
  proj_wmma_kernel<16><<<pgrid, pblk, 0, stream>>>(Wv, bv, x, vws, 0, 1.0f);

  attn_flash_kernel<<<dim3(NSEQ / 64, BATCH), dim3(256), 0, stream>>>(
      qws, kws, vws, x, gamma, (float*)d_out);
}
